// BertOutputWithAdapterFusion_3934190043238
// MI455X (gfx1250) — compile-verified, matrix operand order checked by reading, not run on hardware
//
#include <hip/hip_runtime.h>

// ---------------------------------------------------------------------------
// BertOutputWithAdapterFusion for MI455X (gfx1250, wave32, WMMA bf16)
//
//   h        = hs @ dense_w^T + dense_b                      (WMMA)
//   prenorm  = input + h ; adapter_in = LN(prenorm)
//   query    = prenorm @ query_w^T + query_b                 (WMMA)
//   qk       = query @ key_w          (key GEMM folded away) (WMMA)
//   down[t]  = relu(adapter_in @ down_w[t]^T + down_b[t])    (WMMA, batched)
//   u[t]     = qk @ up_w[t]                                  (WMMA, batched)
//   scores   = <u,down>_D + <qk,up_b[t]> + <query,key_b> ; probs = softmax_t
//   mix      = (probs*down flattened)[N,512] @ up_all[512,H] (WMMA)
//   vin      = mix + probs@up_b + h   (value GEMM folded: sum_t probs = 1)
//   fusion   = vin @ value_w^T                               (WMMA)
//   out      = LN(input + fusion)
// ---------------------------------------------------------------------------

#define TOK 4096   // B*S tokens
#define HDI 1024   // hidden
#define IDI 4096   // intermediate
#define NTK 8      // tasks
#define DDI 64     // bottleneck

typedef __attribute__((ext_vector_type(16))) __bf16 v16bf;
typedef __attribute__((ext_vector_type(8)))  float  v8f;

// --- optional gfx1250 async global->LDS copy path (ASYNCcnt) ---------------
#ifdef __has_builtin
#if __has_builtin(__builtin_amdgcn_global_load_async_to_lds_b128) && \
    __has_builtin(__builtin_amdgcn_s_wait_asynccnt)
#define USE_ASYNC_COPY 1
#endif
#endif

// builtin takes pointers to 16B int vectors (b128); clang adjusts addr space
typedef int v4i32 __attribute__((vector_size(16)));
#define TO_GLB(p) ((v4i32*)(p))
#define TO_LDS(p) ((v4i32*)(p))

// ---------------------------------------------------------------- conversions
__global__ __launch_bounds__(256) void cvt_f32_bf16(
    const float* __restrict__ s, __bf16* __restrict__ d, long n)
{
  long i = ((long)blockIdx.x * 256 + threadIdx.x) * 4;
  if (i + 3 < n) {
    float4 v = *reinterpret_cast<const float4*>(s + i);
    d[i + 0] = (__bf16)v.x; d[i + 1] = (__bf16)v.y;
    d[i + 2] = (__bf16)v.z; d[i + 3] = (__bf16)v.w;
  } else {
    for (; i < n; ++i) d[i] = (__bf16)s[i];
  }
}

// dst[c][r] = src[r][c], fp32 -> bf16 (for key_w used in NN orientation)
__global__ __launch_bounds__(256) void transpose_cvt(
    const float* __restrict__ src, __bf16* __restrict__ dst, int rows, int cols)
{
  __shared__ float tile[32][33];
  const int bx = blockIdx.x * 32, by = blockIdx.y * 32;
  const int tx = threadIdx.x, ty = threadIdx.y;
  #pragma unroll
  for (int j = 0; j < 32; j += 8)
    tile[ty + j][tx] = src[(long)(by + ty + j) * cols + (bx + tx)];
  __syncthreads();
  #pragma unroll
  for (int j = 0; j < 32; j += 8)
    dst[(long)(bx + ty + j) * rows + (by + tx)] = (__bf16)tile[tx][ty + j];
}

// up_w [T,H,D] -> upT [T*D, H] (rows = (t,d), K = h)  and  upmix [H, T*D]
__global__ __launch_bounds__(256) void repack_up(
    const float* __restrict__ upw, __bf16* __restrict__ upT,
    __bf16* __restrict__ upmix)
{
  long i = (long)blockIdx.x * 256 + threadIdx.x;      // over T*H*D
  if (i >= (long)NTK * HDI * DDI) return;
  int t = (int)(i >> 16);                              // H*D = 65536
  int rem = (int)(i & 65535);
  int h = rem >> 6, d = rem & 63;
  __bf16 b = (__bf16)upw[i];
  upT[((long)(t * DDI + d) << 10) + h] = b;            // pitch H
  upmix[((long)h << 9) + t * DDI + d] = b;             // pitch T*D = 512
}

// -------------------------------------------------------------- WMMA NT GEMM
// C[m,j] = sum_k A[m,k] * B[j,k]   (A:[M,K], B:[N,K] row-major bf16)
// 256 threads = 8 waves, wave grid 4x2, wave tile (BM/4)x(BN/2).
// Double-buffered LDS; next tile's global loads overlap the WMMA block.
template<int BM, int BN>
__global__ __launch_bounds__(256) void wmma_gemm_nt(
    const __bf16* __restrict__ A, const __bf16* __restrict__ B,
    float* __restrict__ Cf, __bf16* __restrict__ Cb,
    const float* __restrict__ bias,
    int M, int N, int K,
    long long strideA, long long strideB, long long strideC,
    long long strideBias, int relu)
{
  constexpr int BK = 32;
  constexpr int LP = 56;                 // LDS pitch (bf16) = 112B, 16B aligned
  constexpr int WTM = BM / 4;
  constexpr int WTN = BN / 2;
  constexpr int FM = WTM / 16;
  constexpr int FN = WTN / 16;
  constexpr int AIT = BM * BK / 8 / 256; // 16B chunks per thread (A tile)
  constexpr int BIT = BN * BK / 8 / 256; // 16B chunks per thread (B tile)
  static_assert(AIT >= 1 && BIT >= 1, "tile too small for 256 threads");

  __shared__ __bf16 smA[2][BM * LP];
  __shared__ __bf16 smB[2][BN * LP];

  const int tid = threadIdx.x;
  const int wv = tid >> 5, lane = tid & 31;
  const int wr = wv & 3, wc = wv >> 2;
  const int r = lane & 15, hi = lane >> 4;
  const int bm0 = blockIdx.x * BM, bn0 = blockIdx.y * BN;
  const int bz = blockIdx.z;

  const __bf16* Ab = A + (long long)bz * strideA;
  const __bf16* Bb = B + (long long)bz * strideB;

  // per-thread staging coordinates (hoisted out of the K loop)
  const int srow = tid >> 2, skc = (tid & 3) * 8;   // row within tile, K chunk
  const __bf16* aSrc = Ab + (long long)(bm0 + srow) * K + skc;
  const __bf16* bSrc = Bb + (long long)(bn0 + srow) * K + skc;

  const v8f vzero = {0.f, 0.f, 0.f, 0.f, 0.f, 0.f, 0.f, 0.f};
  v8f acc[FM][FN];
  #pragma unroll
  for (int i = 0; i < FM; ++i)
    #pragma unroll
    for (int j = 0; j < FN; ++j) acc[i][j] = vzero;

#ifdef USE_ASYNC_COPY
  auto stage_async = [&](int k0, int buf) {
    #pragma unroll
    for (int it = 0; it < AIT; ++it)
      __builtin_amdgcn_global_load_async_to_lds_b128(
          TO_GLB(aSrc + (long long)it * 64 * K + k0),
          TO_LDS(&smA[buf][(srow + it * 64) * LP + skc]), 0, 0);
    #pragma unroll
    for (int it = 0; it < BIT; ++it)
      __builtin_amdgcn_global_load_async_to_lds_b128(
          TO_GLB(bSrc + (long long)it * 64 * K + k0),
          TO_LDS(&smB[buf][(srow + it * 64) * LP + skc]), 0, 0);
  };
#else
  uint4 ra[AIT], rb[BIT];
  auto stage_load = [&](int k0) {
    #pragma unroll
    for (int it = 0; it < AIT; ++it)
      ra[it] = *reinterpret_cast<const uint4*>(aSrc + (long long)it * 64 * K + k0);
    #pragma unroll
    for (int it = 0; it < BIT; ++it)
      rb[it] = *reinterpret_cast<const uint4*>(bSrc + (long long)it * 64 * K + k0);
  };
  auto stage_store = [&](int buf) {
    #pragma unroll
    for (int it = 0; it < AIT; ++it)
      *reinterpret_cast<uint4*>(&smA[buf][(srow + it * 64) * LP + skc]) = ra[it];
    #pragma unroll
    for (int it = 0; it < BIT; ++it)
      *reinterpret_cast<uint4*>(&smB[buf][(srow + it * 64) * LP + skc]) = rb[it];
  };
#endif

  // prologue: stage tile 0 into buffer 0
#ifdef USE_ASYNC_COPY
  stage_async(0, 0);
  __builtin_amdgcn_s_wait_asynccnt(0);
#else
  stage_load(0);
  stage_store(0);
#endif
  __syncthreads();

  int cur = 0;
  for (int k0 = 0; k0 < K; k0 += BK) {
    const bool hasNext = (k0 + BK) < K;
    // issue next tile's global loads before computing (overlap)
#ifdef USE_ASYNC_COPY
    if (hasNext) stage_async(k0 + BK, cur ^ 1);
#else
    if (hasNext) stage_load(k0 + BK);
#endif

    // fragments per ISA 16-bit A/B layout:
    // lane<16 holds K{0..7,16..23}, lane>=16 holds K{8..15,24..31}
    v16bf af[FM], bfv[FN];
    #pragma unroll
    for (int i = 0; i < FM; ++i) {
      int row = wr * WTM + i * 16 + r;
      union { uint4 u[2]; v16bf v; } u;
      u.u[0] = *reinterpret_cast<const uint4*>(&smA[cur][row * LP + hi * 8]);
      u.u[1] = *reinterpret_cast<const uint4*>(&smA[cur][row * LP + 16 + hi * 8]);
      af[i] = u.v;
    }
    #pragma unroll
    for (int j = 0; j < FN; ++j) {
      int row = wc * WTN + j * 16 + r;
      union { uint4 u[2]; v16bf v; } u;
      u.u[0] = *reinterpret_cast<const uint4*>(&smB[cur][row * LP + hi * 8]);
      u.u[1] = *reinterpret_cast<const uint4*>(&smB[cur][row * LP + 16 + hi * 8]);
      bfv[j] = u.v;
    }
    #pragma unroll
    for (int i = 0; i < FM; ++i)
      #pragma unroll
      for (int j = 0; j < FN; ++j)
        acc[i][j] = __builtin_amdgcn_wmma_f32_16x16x32_bf16(
            false, af[i], false, bfv[j], (short)0, acc[i][j], false, false);

    // publish next tile into the other buffer
#ifdef USE_ASYNC_COPY
    if (hasNext) __builtin_amdgcn_s_wait_asynccnt(0);
#else
    if (hasNext) stage_store(cur ^ 1);
#endif
    __syncthreads();
    cur ^= 1;
  }

  // epilogue: + bias[col], optional relu, f32 and/or bf16 store.
  // C lane layout: lane<16 -> M = v, lane>=16 -> M = 8+v; N = lane&15.
  const float* biasB = bias ? (bias + (long long)bz * strideBias) : nullptr;
  #pragma unroll
  for (int i = 0; i < FM; ++i) {
    #pragma unroll
    for (int j = 0; j < FN; ++j) {
      int gm = bm0 + wr * WTM + i * 16 + hi * 8;
      int gn = bn0 + wc * WTN + j * 16 + r;
      float bv = biasB ? biasB[gn] : 0.0f;
      #pragma unroll
      for (int v = 0; v < 8; ++v) {
        float x = acc[i][j][v] + bv;
        if (relu) x = fmaxf(x, 0.0f);
        long long idx = (long long)bz * strideC + (long long)(gm + v) * N + gn;
        if (Cf) Cf[idx] = x;
        if (Cb) Cb[idx] = (__bf16)x;
      }
    }
  }
}

// ---------------------------------------------------------------- layernorms
__global__ __launch_bounds__(256) void ln1_kernel(
    const float* __restrict__ inT, const float* __restrict__ hd,
    const float* __restrict__ g, const float* __restrict__ b,
    __bf16* __restrict__ preBf, __bf16* __restrict__ ainBf)
{
  const int n = blockIdx.x, tid = threadIdx.x;
  const int wv = tid >> 5, lane = tid & 31;
  float x[4], s = 0.f, ss = 0.f;
  #pragma unroll
  for (int i = 0; i < 4; ++i) {
    int h = tid + i * 256;
    float v = inT[(long)n * HDI + h] + hd[(long)n * HDI + h];
    x[i] = v; s += v; ss += v * v;
  }
  #pragma unroll
  for (int m = 16; m; m >>= 1) { s += __shfl_xor(s, m, 32); ss += __shfl_xor(ss, m, 32); }
  __shared__ float rs[8], rss[8];
  if (lane == 0) { rs[wv] = s; rss[wv] = ss; }
  __syncthreads();
  s = 0.f; ss = 0.f;
  #pragma unroll
  for (int w = 0; w < 8; ++w) { s += rs[w]; ss += rss[w]; }
  float mean = s * (1.0f / HDI);
  float var  = ss * (1.0f / HDI) - mean * mean;
  float rstd = rsqrtf(var + 1e-12f);
  #pragma unroll
  for (int i = 0; i < 4; ++i) {
    int h = tid + i * 256;
    preBf[(long)n * HDI + h] = (__bf16)x[i];
    ainBf[(long)n * HDI + h] = (__bf16)((x[i] - mean) * rstd * g[h] + b[h]);
  }
}

__global__ __launch_bounds__(256) void ln2_kernel(
    const float* __restrict__ inT, const float* __restrict__ fus,
    const float* __restrict__ g, const float* __restrict__ b,
    float* __restrict__ out)
{
  const int n = blockIdx.x, tid = threadIdx.x;
  const int wv = tid >> 5, lane = tid & 31;
  float x[4], s = 0.f, ss = 0.f;
  #pragma unroll
  for (int i = 0; i < 4; ++i) {
    int h = tid + i * 256;
    float v = inT[(long)n * HDI + h] + fus[(long)n * HDI + h];
    x[i] = v; s += v; ss += v * v;
  }
  #pragma unroll
  for (int m = 16; m; m >>= 1) { s += __shfl_xor(s, m, 32); ss += __shfl_xor(ss, m, 32); }
  __shared__ float rs[8], rss[8];
  if (lane == 0) { rs[wv] = s; rss[wv] = ss; }
  __syncthreads();
  s = 0.f; ss = 0.f;
  #pragma unroll
  for (int w = 0; w < 8; ++w) { s += rs[w]; ss += rss[w]; }
  float mean = s * (1.0f / HDI);
  float var  = ss * (1.0f / HDI) - mean * mean;
  float rstd = rsqrtf(var + 1e-12f);
  #pragma unroll
  for (int i = 0; i < 4; ++i) {
    int h = tid + i * 256;
    out[(long)n * HDI + h] = (x[i] - mean) * rstd * g[h] + b[h];
  }
}

// ----------------------------------------------- scores + softmax over tasks
// scores[t,n] = <u[t,n],down[t,n]>_D + <qk[n],up_b[t]>_H + <query[n],key_b>_H
__global__ __launch_bounds__(256) void score_softmax_kernel(
    const __bf16* __restrict__ qk, const __bf16* __restrict__ query,
    const float* __restrict__ key_b, const float* __restrict__ up_b,
    const float* __restrict__ u, const float* __restrict__ down,
    float* __restrict__ probs)
{
  const int n = blockIdx.x, tid = threadIdx.x;
  const int wv = tid >> 5, lane = tid & 31;

  float pq[NTK] = {0.f, 0.f, 0.f, 0.f, 0.f, 0.f, 0.f, 0.f};
  float pc = 0.f;
  for (int h = tid; h < HDI; h += 256) {
    float q  = (float)qk[(long)n * HDI + h];
    float qr = (float)query[(long)n * HDI + h];
    pc += qr * key_b[h];
    #pragma unroll
    for (int t = 0; t < NTK; ++t) pq[t] += q * up_b[t * HDI + h];
  }
  #pragma unroll
  for (int m = 16; m; m >>= 1) {
    pc += __shfl_xor(pc, m, 32);
    #pragma unroll
    for (int t = 0; t < NTK; ++t) pq[t] += __shfl_xor(pq[t], m, 32);
  }
  __shared__ float sp[8][NTK + 1];
  __shared__ float sc[8];
  __shared__ float sd[8];
  if (lane == 0) {
    #pragma unroll
    for (int t = 0; t < NTK; ++t) sp[wv][t] = pq[t];
    sc[wv] = pc;
  }
  {   // wave wv owns task t = wv : <u,down> over D=64
    const float* up = u    + ((long)wv * TOK + n) * DDI;
    const float* dp = down + ((long)wv * TOK + n) * DDI;
    float p = up[lane] * dp[lane] + up[lane + 32] * dp[lane + 32];
    #pragma unroll
    for (int m = 16; m; m >>= 1) p += __shfl_xor(p, m, 32);
    if (lane == 0) sd[wv] = p;
  }
  __syncthreads();
  if (tid == 0) {
    float c0 = 0.f;
    #pragma unroll
    for (int w = 0; w < 8; ++w) c0 += sc[w];
    float s[NTK], mx = -1e30f;
    #pragma unroll
    for (int t = 0; t < NTK; ++t) {
      float a = 0.f;
      #pragma unroll
      for (int w = 0; w < 8; ++w) a += sp[w][t];
      s[t] = sd[t] + a + c0;
      mx = fmaxf(mx, s[t]);
    }
    float den = 0.f;
    #pragma unroll
    for (int t = 0; t < NTK; ++t) { s[t] = __expf(s[t] - mx); den += s[t]; }
    float inv = 1.0f / den;
    #pragma unroll
    for (int t = 0; t < NTK; ++t) probs[(long)n * NTK + t] = s[t] * inv;
  }
}

// wdown[n, t*64+d] = bf16(probs[n,t] * down[t,n,d])
__global__ __launch_bounds__(256) void wdown_kernel(
    const float* __restrict__ down, const float* __restrict__ probs,
    __bf16* __restrict__ wd)
{
  long i = (long)blockIdx.x * 256 + threadIdx.x;   // over N * T * D
  if (i >= (long)TOK * NTK * DDI) return;
  int n = (int)(i >> 9);
  int td = (int)(i & 511);
  int t = td >> 6, d = td & 63;
  wd[i] = (__bf16)(probs[(long)n * NTK + t] *
                   down[((long)t * TOK + n) * DDI + d]);
}

// vin = bf16( mix + h_dense + sum_t probs[t]*up_b[t] )
__global__ __launch_bounds__(256) void vin_kernel(
    const float* __restrict__ mix, const float* __restrict__ hd,
    const float* __restrict__ up_b, const float* __restrict__ probs,
    __bf16* __restrict__ vin)
{
  long i = (long)blockIdx.x * 256 + threadIdx.x;   // over N * H
  if (i >= (long)TOK * HDI) return;
  int n = (int)(i >> 10), h = (int)(i & 1023);
  float acc = mix[i] + hd[i];
  #pragma unroll
  for (int t = 0; t < NTK; ++t)
    acc += probs[(long)n * NTK + t] * up_b[t * HDI + h];
  vin[i] = (__bf16)acc;
}

// ---------------------------------------------------------------------------
extern "C" void kernel_launch(void* const* d_in, const int* in_sizes, int n_in,
                              void* d_out, int out_size, void* d_ws, size_t ws_size,
                              hipStream_t stream) {
  (void)in_sizes; (void)n_in; (void)out_size; (void)ws_size;
  const float* hs  = (const float*)d_in[0];
  const float* inT = (const float*)d_in[1];
  const float* dw  = (const float*)d_in[2];
  const float* db  = (const float*)d_in[3];
  const float* lng = (const float*)d_in[4];
  const float* lnb = (const float*)d_in[5];
  const float* dnw = (const float*)d_in[6];
  const float* dnb = (const float*)d_in[7];
  const float* upw = (const float*)d_in[8];
  const float* upb = (const float*)d_in[9];
  const float* kw  = (const float*)d_in[10];
  const float* kb  = (const float*)d_in[11];
  const float* qw  = (const float*)d_in[12];
  const float* qb  = (const float*)d_in[13];
  const float* vw  = (const float*)d_in[14];
  float* out = (float*)d_out;

  char* ws = (char*)d_ws;
  size_t off = 0;
  auto alloc = [&](size_t bytes) -> char* {
    char* p = ws + off;
    off += (bytes + 255) & ~(size_t)255;
    return p;
  };

  __bf16* hs_bf   = (__bf16*)alloc((size_t)TOK * IDI * 2);
  __bf16* dw_bf   = (__bf16*)alloc((size_t)HDI * IDI * 2);
  float*  hdense  = (float*) alloc((size_t)TOK * HDI * 4);
  __bf16* pre_bf  = (__bf16*)alloc((size_t)TOK * HDI * 2);
  __bf16* ain_bf  = (__bf16*)alloc((size_t)TOK * HDI * 2);
  __bf16* qw_bf   = (__bf16*)alloc((size_t)HDI * HDI * 2);
  __bf16* kwT_bf  = (__bf16*)alloc((size_t)HDI * HDI * 2);
  __bf16* vw_bf   = (__bf16*)alloc((size_t)HDI * HDI * 2);
  __bf16* dnw_bf  = (__bf16*)alloc((size_t)NTK * DDI * HDI * 2);
  __bf16* upT_bf  = (__bf16*)alloc((size_t)NTK * DDI * HDI * 2);
  __bf16* upmx_bf = (__bf16*)alloc((size_t)NTK * DDI * HDI * 2);
  __bf16* query_bf= (__bf16*)alloc((size_t)TOK * HDI * 2);
  __bf16* qk_bf   = (__bf16*)alloc((size_t)TOK * HDI * 2);
  float*  down_f  = (float*) alloc((size_t)NTK * TOK * DDI * 4);
  float*  u_f     = (float*) alloc((size_t)NTK * TOK * DDI * 4);
  float*  probs   = (float*) alloc((size_t)TOK * NTK * 4);
  __bf16* wdown_bf= (__bf16*)alloc((size_t)TOK * NTK * DDI * 2);
  float*  mix_f   = (float*) alloc((size_t)TOK * HDI * 4);
  __bf16* vin_bf  = (__bf16*)alloc((size_t)TOK * HDI * 2);
  float*  fus_f   = (float*) alloc((size_t)TOK * HDI * 4);

  // ---- stage 0: precision conversion / weight repacking
  {
    long n;
    n = (long)TOK * IDI;
    cvt_f32_bf16<<<dim3((unsigned)((n + 1023) / 1024)), dim3(256), 0, stream>>>(hs, hs_bf, n);
    n = (long)HDI * IDI;
    cvt_f32_bf16<<<dim3((unsigned)((n + 1023) / 1024)), dim3(256), 0, stream>>>(dw, dw_bf, n);
    n = (long)HDI * HDI;
    cvt_f32_bf16<<<dim3((unsigned)((n + 1023) / 1024)), dim3(256), 0, stream>>>(qw, qw_bf, n);
    cvt_f32_bf16<<<dim3((unsigned)((n + 1023) / 1024)), dim3(256), 0, stream>>>(vw, vw_bf, n);
    n = (long)NTK * DDI * HDI;
    cvt_f32_bf16<<<dim3((unsigned)((n + 1023) / 1024)), dim3(256), 0, stream>>>(dnw, dnw_bf, n);
    transpose_cvt<<<dim3(HDI / 32, HDI / 32), dim3(32, 8), 0, stream>>>(kw, kwT_bf, HDI, HDI);
    repack_up<<<dim3((NTK * HDI * DDI) / 256), dim3(256), 0, stream>>>(upw, upT_bf, upmx_bf);
  }

  // ---- G1: h = hs @ dense_w^T + dense_b
  wmma_gemm_nt<128, 128><<<dim3(TOK / 128, HDI / 128, 1), dim3(256), 0, stream>>>(
      hs_bf, dw_bf, hdense, nullptr, db, TOK, HDI, IDI, 0, 0, 0, 0, 0);

  // ---- LN1: prenorm(bf16) + adapter_in(bf16)
  ln1_kernel<<<dim3(TOK), dim3(256), 0, stream>>>(inT, hdense, lng, lnb, pre_bf, ain_bf);

  // ---- G2: query = prenorm @ query_w^T + query_b  (bf16 out)
  wmma_gemm_nt<128, 128><<<dim3(TOK / 128, HDI / 128, 1), dim3(256), 0, stream>>>(
      pre_bf, qw_bf, nullptr, query_bf, qb, TOK, HDI, HDI, 0, 0, 0, 0, 0);

  // ---- G3: qk = query @ key_w  (bf16 out)
  wmma_gemm_nt<128, 128><<<dim3(TOK / 128, HDI / 128, 1), dim3(256), 0, stream>>>(
      query_bf, kwT_bf, nullptr, qk_bf, nullptr, TOK, HDI, HDI, 0, 0, 0, 0, 0);

  // ---- G4: down[t] = relu(adapter_in @ down_w[t]^T + down_b[t])  (batched over t)
  wmma_gemm_nt<128, 64><<<dim3(TOK / 128, 1, NTK), dim3(256), 0, stream>>>(
      ain_bf, dnw_bf, down_f, nullptr, dnb, TOK, DDI, HDI,
      0, (long long)DDI * HDI, (long long)TOK * DDI, DDI, 1);

  // ---- G5: u[t] = qk @ up_w[t]  (batched over t)
  wmma_gemm_nt<128, 64><<<dim3(TOK / 128, 1, NTK), dim3(256), 0, stream>>>(
      qk_bf, upT_bf, u_f, nullptr, nullptr, TOK, DDI, HDI,
      0, (long long)DDI * HDI, (long long)TOK * DDI, 0, 0);

  // ---- scores + softmax over tasks
  score_softmax_kernel<<<dim3(TOK), dim3(256), 0, stream>>>(
      qk_bf, query_bf, kb, upb, u_f, down_f, probs);

  // ---- probs-weighted down, flattened [N, T*D]
  wdown_kernel<<<dim3((TOK * NTK * DDI) / 256), dim3(256), 0, stream>>>(
      down_f, probs, wdown_bf);

  // ---- G6: mix = wdown @ up_all^T   ([4096,512] x [1024,512])
  wmma_gemm_nt<128, 128><<<dim3(TOK / 128, HDI / 128, 1), dim3(256), 0, stream>>>(
      wdown_bf, upmx_bf, mix_f, nullptr, nullptr, TOK, HDI, NTK * DDI, 0, 0, 0, 0, 0);

  // ---- vin = mix + h + probs@up_b  (bf16)
  vin_kernel<<<dim3((TOK * HDI) / 256), dim3(256), 0, stream>>>(
      mix_f, hdense, upb, probs, vin_bf);

  // ---- G7: fusion = vin @ value_w^T
  wmma_gemm_nt<128, 128><<<dim3(TOK / 128, HDI / 128, 1), dim3(256), 0, stream>>>(
      vin_bf, vw_bf, fus_f, nullptr, nullptr, TOK, HDI, HDI, 0, 0, 0, 0, 0);

  // ---- LN2 -> out
  ln2_kernel<<<dim3(TOK), dim3(256), 0, stream>>>(inT, fus_f, lng, lnb, out);
}
